// FasterRCNN_61263413510696
// MI455X (gfx1250) — compile-verified, matrix operand order checked by reading, not run on hardware
//
#include <hip/hip_runtime.h>
#include <hip/hip_bf16.h>

typedef __attribute__((ext_vector_type(16))) _Float16 v16h;
typedef __attribute__((ext_vector_type(8)))  _Float16 v8h;
typedef __attribute__((ext_vector_type(8)))  float    v8f;
typedef int v4i_vec __attribute__((vector_size(16)));

#define IMG_SZ 512
#define FM 32
#define NPOS 1024          // 32*32 feature positions
#define CF 512             // feature channels
#define KA 9               // anchors per position
#define NANCH 9216         // 1024*9
#define NPAD 16384         // sort padding (pow2 >= NANCH)
#define IOU_THR 0.7f
#define TOPK 300

#if __has_builtin(__builtin_amdgcn_global_load_async_to_lds_b128)
#define HAVE_ASYNC_LDS 1
#else
#define HAVE_ASYNC_LDS 0
#endif

__device__ __forceinline__ void wait_async0() {
#if __has_builtin(__builtin_amdgcn_s_wait_asynccnt)
  __builtin_amdgcn_s_wait_asynccnt(0);
#else
  asm volatile("s_wait_asynccnt 0" ::: "memory");
#endif
}

// ---------------------------------------------------------------------------
// Fragment loader matching CDNA5 16-bit A/B operand layout for 16x16x32:
// lane L: row/col = L&15, khalf = L>>4.
// f16 slots 0..7  <- K = khalf*8 + s        (contiguous 16B)
// f16 slots 8..15 <- K = 16 + khalf*8 + s-8 (contiguous 16B)
// ---------------------------------------------------------------------------
__device__ __forceinline__ v16h ld_frag(const _Float16* __restrict__ p, int khalf) {
  const v8h a = *reinterpret_cast<const v8h*>(p + khalf * 8);
  const v8h b = *reinterpret_cast<const v8h*>(p + 16 + khalf * 8);
  v16h r;
#pragma unroll
  for (int i = 0; i < 8; ++i) { r[i] = a[i]; r[i + 8] = b[i]; }
  return r;
}

// ---------------------------------------------------------------------------
// Generic WMMA GEMM: C = act(A[MxK] * Bt[NxK]^T + bias), one wave per 16x16 tile
// (used for stage A patchify GEMM and the fused 1x1 head)
// ---------------------------------------------------------------------------
__global__ void gemm_wmma(const _Float16* __restrict__ A, const _Float16* __restrict__ Bt,
                          const float* __restrict__ bias, _Float16* __restrict__ Ch,
                          float* __restrict__ Cf, int M, int N, int K, int relu) {
  int wave = (blockIdx.x * blockDim.x + threadIdx.x) >> 5;
  int lane = threadIdx.x & 31;
  int nt = N >> 4;
  int tiles = (M >> 4) * nt;
  if (wave >= tiles) return;
  int tm = wave / nt, tn = wave % nt;
  int r = lane & 15, kh = lane >> 4;
  const _Float16* arow = A + (size_t)(tm * 16 + r) * K;
  const _Float16* brow = Bt + (size_t)(tn * 16 + r) * K;
  v8f acc = {};
  for (int k0 = 0; k0 < K; k0 += 32) {
    v16h af = ld_frag(arow + k0, kh);
    v16h bf = ld_frag(brow + k0, kh);
    acc = __builtin_amdgcn_wmma_f32_16x16x32_f16(false, af, false, bf, (short)0, acc,
                                                 false, false);
  }
  int n = tn * 16 + r;
  float bv = bias ? bias[n] : 0.f;
#pragma unroll
  for (int i = 0; i < 8; ++i) {
    int m = tm * 16 + i + 8 * kh;   // C/D layout: vgpr i -> row i + 8*(lane>>4)
    float v = acc[i] + bv;
    if (relu) v = fmaxf(v, 0.f);
    if (Ch) Ch[(size_t)m * N + n] = (_Float16)v;
    if (Cf) Cf[(size_t)m * N + n] = v;
  }
}

// ---------------------------------------------------------------------------
// 3x3 SAME conv, LDS-staged weights + 4x M register blocking.
// One block (16 waves, 512 thr) per 16-channel output tile; the full weight
// panel [16][9*512] f16 = 147 KB is staged into LDS (async-to-LDS when the
// toolchain exposes it; ASYNCcnt drained before the barrier). Each wave owns
// 4 M-tiles and reuses every LDS B-fragment across 4 WMMAs.
// Weights are read from global exactly once across the whole grid.
// ---------------------------------------------------------------------------
__global__ void __launch_bounds__(512) conv3_wmma(const _Float16* __restrict__ feat,
                                                  const _Float16* __restrict__ Wt,
                                                  const float* __restrict__ bias,
                                                  _Float16* __restrict__ hid) {
  extern __shared__ char dyn_smem[];
  _Float16* Bs = (_Float16*)dyn_smem;           // [16][4608]
  const int tn = blockIdx.x;                    // 0..31 (N tile)
  const int tid = threadIdx.x;

  const _Float16* wsrc = Wt + (size_t)tn * 16 * (9 * CF);
  const int n16chunks = (16 * 9 * CF * 2) / 16; // 9216 x 16B
#if HAVE_ASYNC_LDS
  {
    const char* g = (const char*)wsrc;
    char* l = dyn_smem;
    for (int c = tid; c < n16chunks; c += 512) {
      __builtin_amdgcn_global_load_async_to_lds_b128(
          (v4i_vec*)(g + (size_t)c * 16), (v4i_vec*)(l + (size_t)c * 16), 0, 0);
    }
    wait_async0();
  }
#else
  for (int c = tid; c < n16chunks; c += 512) {
    ((v8h*)Bs)[c] = ((const v8h*)wsrc)[c];
  }
#endif
  __syncthreads();

  const int wave = tid >> 5, lane = tid & 31;
  const int r = lane & 15, kh = lane >> 4;
  const int n = tn * 16 + r;
  const _Float16* brow = Bs + (size_t)r * (9 * CF);

  v8f z8 = {};
  v8f acc[4];
#pragma unroll
  for (int q = 0; q < 4; ++q) acc[q] = z8;
  v16h zero = {};

  for (int t = 0; t < 9; ++t) {
    const int dy = t / 3 - 1, dx = t % 3 - 1;
    const _Float16* arow[4];
    bool valid[4];
#pragma unroll
    for (int q = 0; q < 4; ++q) {
      int p = (wave * 4 + q) * 16 + r;
      int yy = (p >> 5) + dy, xx = (p & 31) + dx;
      valid[q] = (yy >= 0) && (yy < FM) && (xx >= 0) && (xx < FM);
      int yc = valid[q] ? yy : 0, xc = valid[q] ? xx : 0;
      arow[q] = feat + (size_t)(yc * FM + xc) * CF;
    }
    for (int c0 = 0; c0 < CF; c0 += 32) {
      v16h bf = ld_frag(brow + t * CF + c0, kh);   // ds_load_b128 x2
#pragma unroll
      for (int q = 0; q < 4; ++q) {
        v16h af = ld_frag(arow[q] + c0, kh);
        if (!valid[q]) af = zero;
        acc[q] = __builtin_amdgcn_wmma_f32_16x16x32_f16(false, af, false, bf, (short)0,
                                                        acc[q], false, false);
      }
    }
  }
  float bv = bias[n];
#pragma unroll
  for (int q = 0; q < 4; ++q) {
#pragma unroll
    for (int i = 0; i < 8; ++i) {
      int m = (wave * 4 + q) * 16 + i + 8 * kh;
      hid[(size_t)m * CF + n] = (_Float16)fmaxf(acc[q][i] + bv, 0.f);
    }
  }
}

// ---------------------------------------------------------------------------
// Data-prep kernels (f32 -> f16 conversions / layout permutes)
// ---------------------------------------------------------------------------
__global__ void patches_kernel(const float* __restrict__ img, _Float16* __restrict__ Ap) {
  int e = blockIdx.x * blockDim.x + threadIdx.x;
  if (e >= NPOS * 768) return;
  int pos = e / 768, k = e % 768;
  int c = k >> 8, rem = k & 255, ii = rem >> 4, jj = rem & 15;
  int y = pos >> 5, x = pos & 31;
  Ap[e] = (_Float16)img[(size_t)c * IMG_SZ * IMG_SZ + (size_t)(y * 16 + ii) * IMG_SZ +
                        (x * 16 + jj)];
}

__global__ void cvt_kernel(const float* __restrict__ s, _Float16* __restrict__ d, int n) {
  int e = blockIdx.x * blockDim.x + threadIdx.x;
  if (e < n) d[e] = (_Float16)s[e];
}

// w_rpn OIHW [n][c][3][3] -> Wt [n][t*512+c]
__global__ void wrpn_kernel(const float* __restrict__ w, _Float16* __restrict__ Wt) {
  int e = blockIdx.x * blockDim.x + threadIdx.x;
  if (e >= CF * CF * 9) return;
  int n = e / (CF * 9), r2 = e % (CF * 9), t = r2 / CF, c = r2 % CF;
  Wt[e] = (_Float16)w[((size_t)n * CF + c) * 9 + t];
}

// fused head weights: rows 0..17 = w_cls, 18..53 = w_box, 54..63 = 0
__global__ void whead_kernel(const float* __restrict__ wc, const float* __restrict__ wb,
                             _Float16* __restrict__ Wt) {
  int e = blockIdx.x * blockDim.x + threadIdx.x;
  if (e >= 64 * CF) return;
  int n = e / CF, c = e % CF;
  float v = (n < 18) ? wc[n * CF + c] : ((n < 54) ? wb[(n - 18) * CF + c] : 0.f);
  Wt[e] = (_Float16)v;
}

// ---------------------------------------------------------------------------
// Anchor decode + clip; anchor a = pos*9 + k; score = headOut[pos][9+k]
// ---------------------------------------------------------------------------
__global__ void decode_kernel(const float* __restrict__ headOut, const float* __restrict__ anchors,
                              const float* __restrict__ b_cls, const float* __restrict__ b_box,
                              float* __restrict__ scores, float* __restrict__ props) {
  int a = blockIdx.x * blockDim.x + threadIdx.x;
  if (a >= NANCH) return;
  int pos = a / KA, k = a % KA;
  const float* row = headOut + (size_t)pos * 64;
  scores[a] = row[9 + k] + b_cls[9 + k];
  float dx = row[18 + k * 4 + 0] + b_box[k * 4 + 0];
  float dy = row[18 + k * 4 + 1] + b_box[k * 4 + 1];
  float dw = fminf(fmaxf(row[18 + k * 4 + 2] + b_box[k * 4 + 2], -4.f), 4.f);
  float dh = fminf(fmaxf(row[18 + k * 4 + 3] + b_box[k * 4 + 3], -4.f), 4.f);
  float ax1 = anchors[a * 4 + 0], ay1 = anchors[a * 4 + 1];
  float ax2 = anchors[a * 4 + 2], ay2 = anchors[a * 4 + 3];
  float wa = ax2 - ax1, ha = ay2 - ay1;
  float cx = ax1 + 0.5f * wa + dx * wa;
  float cy = ay1 + 0.5f * ha + dy * ha;
  float w = wa * expf(dw), h = ha * expf(dh);
  props[a * 4 + 0] = fmaxf(cx - 0.5f * w, 0.f);
  props[a * 4 + 1] = fmaxf(cy - 0.5f * h, 0.f);
  props[a * 4 + 2] = fminf(cx + 0.5f * w, (float)(IMG_SZ - 1));
  props[a * 4 + 3] = fminf(cy + 0.5f * h, (float)(IMG_SZ - 1));
}

// ---------------------------------------------------------------------------
// Single-workgroup bitonic sort, 16384 u64 keys in 128 KB LDS (CDNA5: 320KB/WG)
// key = (~monotone(score) << 32) | idx  -> ascending == stable argsort(-score)
// ---------------------------------------------------------------------------
__global__ void sort_kernel(const float* __restrict__ scores, const float* __restrict__ props,
                            float* __restrict__ boxes_s, int* __restrict__ order) {
  extern __shared__ char dyn_smem[];
  unsigned long long* key = (unsigned long long*)dyn_smem;
  const int NT = blockDim.x;
  for (int i = threadIdx.x; i < NPAD; i += NT) {
    unsigned long long kk;
    if (i < NANCH) {
      unsigned b = __float_as_uint(scores[i]);
      b = (b & 0x80000000u) ? ~b : (b | 0x80000000u);  // ascending-monotone map
      kk = ((unsigned long long)(~b) << 32) | (unsigned)i;
    } else {
      kk = ~0ull;
    }
    key[i] = kk;
  }
  __syncthreads();
  for (int k = 2; k <= NPAD; k <<= 1) {
    for (int j = k >> 1; j > 0; j >>= 1) {
      for (int i = threadIdx.x; i < NPAD; i += NT) {
        int ixj = i ^ j;
        if (ixj > i) {
          bool up = ((i & k) == 0);
          unsigned long long a = key[i], b = key[ixj];
          if ((a > b) == up) { key[i] = b; key[ixj] = a; }
        }
      }
      __syncthreads();
    }
  }
  for (int i = threadIdx.x; i < NANCH; i += NT) {
    int idx = (int)(unsigned)(key[i] & 0xFFFFFFFFull);
    order[i] = idx;
    boxes_s[i * 4 + 0] = props[idx * 4 + 0];
    boxes_s[i * 4 + 1] = props[idx * 4 + 1];
    boxes_s[i * 4 + 2] = props[idx * 4 + 2];
    boxes_s[i * 4 + 3] = props[idx * 4 + 3];
  }
}

// ---------------------------------------------------------------------------
// Single-workgroup greedy NMS; entire 9216-box working set + flags in ~184KB LDS
// ---------------------------------------------------------------------------
__global__ void nms_kernel(const float* __restrict__ boxes_s, int* __restrict__ keep_g) {
  extern __shared__ char dyn_smem[];
  float4* box = (float4*)dyn_smem;
  int* keep = (int*)(dyn_smem + (size_t)NANCH * sizeof(float4));
  const int NT = blockDim.x;
  for (int i = threadIdx.x; i < NANCH; i += NT) {
    box[i] = make_float4(boxes_s[i * 4 + 0], boxes_s[i * 4 + 1], boxes_s[i * 4 + 2],
                         boxes_s[i * 4 + 3]);
    keep[i] = 1;
  }
  __syncthreads();
  for (int i = 0; i < NANCH - 1; ++i) {
    if (keep[i]) {                       // uniform read: all lanes see same LDS value
      float4 bi = box[i];
      float ai = (bi.z - bi.x) * (bi.w - bi.y);
      for (int j = i + 1 + threadIdx.x; j < NANCH; j += NT) {
        if (!keep[j]) continue;
        float4 bj = box[j];
        float aj = (bj.z - bj.x) * (bj.w - bj.y);
        float iw = fmaxf(fminf(bi.z, bj.z) - fmaxf(bi.x, bj.x), 0.f);
        float ih = fmaxf(fminf(bi.w, bj.w) - fmaxf(bi.y, bj.y), 0.f);
        float inter = iw * ih;
        float iou = inter / (ai + aj - inter + 1e-9f);
        if (iou > IOU_THR) keep[j] = 0;
      }
    }
    __syncthreads();
  }
  for (int i = threadIdx.x; i < NANCH; i += NT) keep_g[i] = keep[i];
}

// ---------------------------------------------------------------------------
// Stable selection == argsort((~keep).astype(int), stable)[:300]
// rank(i) = keep[i] ? keptBefore(i) : total_kept + (i - keptBefore(i))
// ---------------------------------------------------------------------------
__global__ void select_kernel(const float* __restrict__ boxes_s, const int* __restrict__ keep,
                              float* __restrict__ out) {
  __shared__ int cnt[1024];
  __shared__ int totalSh;
  int tid = threadIdx.x;
  const int CHUNK = NANCH / 1024;  // 9
  int base = tid * CHUNK;
  int c0 = 0;
  for (int q = 0; q < CHUNK; ++q) c0 += keep[base + q];
  cnt[tid] = c0;
  __syncthreads();
  for (int off = 1; off < 1024; off <<= 1) {   // inclusive Hillis-Steele scan
    int v = (tid >= off) ? cnt[tid - off] : 0;
    __syncthreads();
    cnt[tid] += v;
    __syncthreads();
  }
  if (tid == 1023) totalSh = cnt[1023];
  __syncthreads();
  int total = totalSh;
  int keptBefore = cnt[tid] - c0;              // exclusive prefix of kept
  for (int q = 0; q < CHUNK; ++q) {
    int i = base + q;
    int rank;
    if (keep[i]) { rank = keptBefore; keptBefore++; }
    else         { rank = total + (i - keptBefore); }
    if (rank < TOPK) {
      out[rank * 4 + 0] = boxes_s[i * 4 + 0];
      out[rank * 4 + 1] = boxes_s[i * 4 + 1];
      out[rank * 4 + 2] = boxes_s[i * 4 + 2];
      out[rank * 4 + 3] = boxes_s[i * 4 + 3];
    }
  }
}

// ---------------------------------------------------------------------------
extern "C" void kernel_launch(void* const* d_in, const int* in_sizes, int n_in,
                              void* d_out, int out_size, void* d_ws, size_t ws_size,
                              hipStream_t stream) {
  const float* img   = (const float*)d_in[0];
  const float* anch  = (const float*)d_in[1];
  const float* w_bb  = (const float*)d_in[2];
  const float* b_bb  = (const float*)d_in[3];
  const float* w_rpn = (const float*)d_in[4];
  const float* b_rpn = (const float*)d_in[5];
  const float* w_cls = (const float*)d_in[6];
  const float* b_cls = (const float*)d_in[7];
  const float* w_box = (const float*)d_in[8];
  const float* b_box = (const float*)d_in[9];
  float* out = (float*)d_out;

  char* ws = (char*)d_ws;
  size_t off = 0;
  auto alloc = [&](size_t bytes) -> void* {
    void* p = ws + off;
    off += (bytes + 255) & ~(size_t)255;
    return p;
  };
  _Float16* Ap      = (_Float16*)alloc((size_t)NPOS * 768 * 2);   // im2col patches
  _Float16* WbbT    = (_Float16*)alloc((size_t)CF * 768 * 2);     // backbone weights f16
  _Float16* feat    = (_Float16*)alloc((size_t)NPOS * CF * 2);
  _Float16* Wrpn    = (_Float16*)alloc((size_t)CF * 9 * CF * 2);  // [n][t*512+c]
  _Float16* hid     = (_Float16*)alloc((size_t)NPOS * CF * 2);
  _Float16* Whead   = (_Float16*)alloc((size_t)64 * CF * 2);
  float*    headOut = (float*)alloc((size_t)NPOS * 64 * 4);
  float*    scores  = (float*)alloc((size_t)NANCH * 4);
  float*    props   = (float*)alloc((size_t)NANCH * 4 * 4);
  float*    boxes_s = (float*)alloc((size_t)NANCH * 4 * 4);
  int*      keep    = (int*)alloc((size_t)NANCH * 4);
  int*      order   = (int*)alloc((size_t)NANCH * 4);
  (void)order; (void)in_sizes; (void)n_in; (void)out_size; (void)ws_size;

  const int T = 256;
  // data prep
  patches_kernel<<<(NPOS * 768 + T - 1) / T, T, 0, stream>>>(img, Ap);
  cvt_kernel<<<(CF * 768 + T - 1) / T, T, 0, stream>>>(w_bb, WbbT, CF * 768);
  wrpn_kernel<<<(CF * CF * 9 + T - 1) / T, T, 0, stream>>>(w_rpn, Wrpn);
  whead_kernel<<<(64 * CF + T - 1) / T, T, 0, stream>>>(w_cls, w_box, Whead);

  // stage A: feat = relu(Ap[1024x768] * WbbT^T + b_bb)  -> 2048 tiles, 8 waves/blk
  gemm_wmma<<<2048 / 8, 256, 0, stream>>>(Ap, WbbT, b_bb, feat, nullptr, NPOS, CF, 768, 1);
  // stage B: LDS-staged weights (147KB/WG), 16 waves/blk, 4 M-tiles per wave
  conv3_wmma<<<32, 512, (size_t)16 * 9 * CF * 2, stream>>>(feat, Wrpn, b_rpn, hid);
  // stage C: fused cls+box head, N padded 54->64        -> 256 tiles
  gemm_wmma<<<256 / 8, 256, 0, stream>>>(hid, Whead, nullptr, nullptr, headOut, NPOS, 64, CF, 0);

  decode_kernel<<<(NANCH + T - 1) / T, T, 0, stream>>>(headOut, anch, b_cls, b_box,
                                                       scores, props);
  sort_kernel<<<1, 1024, (size_t)NPAD * 8, stream>>>(scores, props, boxes_s, order);
  nms_kernel<<<1, 1024, (size_t)NANCH * sizeof(float4) + (size_t)NANCH * 4, stream>>>(boxes_s,
                                                                                      keep);
  select_kernel<<<1, 1024, 0, stream>>>(boxes_s, keep, out);
}
